// DbrxFFNQKVSplit_53352083751565
// MI455X (gfx1250) — matrix operand order repaired and checked by reading, not compiled, not run on hardware
//
#include <hip/hip_runtime.h>
#include <hip/hip_bf16.h>

// ---------------------------------------------------------------------------
// DBRX MoE FFN for gfx1250 (MI455X): router + dense-over-experts gated MLP.
// bf16 WMMA (v_wmma_f32_16x16x32_bf16, wave32), f32 accumulation.
// Block tile 64(tokens) x 128(cols); 8 waves, each owns a 16-wide N strip and
// 4 M sub-tiles -> B fragment loaded once per wave per K panel, reused 4x.
// A: row-major LDS [m][k] -> 2x ds_load_b128 per fragment.
// B: transposed LDS [n][k] -> 2x ds_load_b128 per fragment.
// Staging is straight-line (compile-time per-thread assignment), b128 global
// loads + packed bf16 converts; next-panel global_prefetch.
// ---------------------------------------------------------------------------

#define HID   2048
#define FFN   4096
#define NEXP  16
#define NTOK  2048
#define TOPK  4

#define TM    64     // block tile M (tokens)
#define TN    128    // block tile N (cols)
#define KC    32     // K panel per WMMA step

typedef __attribute__((ext_vector_type(4)))  __bf16 bf16x4;
typedef __attribute__((ext_vector_type(8)))  __bf16 bf16x8;
typedef __attribute__((ext_vector_type(16))) __bf16 v16bf;
typedef __attribute__((ext_vector_type(8)))  float  v8f;
typedef __attribute__((ext_vector_type(4)))  float  f32x4;
typedef __bf16 bf16;

// A fragment (16x32 bf16, M x K): row-major LDS row, two contiguous 16B runs.
__device__ __forceinline__ v16bf frag_a(const bf16* row, int half) {
    const bf16x8 lo = *reinterpret_cast<const bf16x8*>(row + half * 8);
    const bf16x8 hi = *reinterpret_cast<const bf16x8*>(row + 16 + half * 8);
    return __builtin_shufflevector(lo, hi, 0, 1, 2, 3, 4, 5, 6, 7,
                                           8, 9, 10, 11, 12, 13, 14, 15);
}

// B fragment (32x16 bf16, K x N): transposed LDS row [n][k], two contiguous 16B runs.
__device__ __forceinline__ v16bf frag_b(const bf16* rowT, int half) {
    const bf16x8 lo = *reinterpret_cast<const bf16x8*>(rowT + half * 8);
    const bf16x8 hi = *reinterpret_cast<const bf16x8*>(rowT + 16 + half * 8);
    return __builtin_shufflevector(lo, hi, 0, 1, 2, 3, 4, 5, 6, 7,
                                           8, 9, 10, 11, 12, 13, 14, 15);
}

__device__ __forceinline__ v8f v8f_zero() {
    return (v8f){0.f, 0.f, 0.f, 0.f, 0.f, 0.f, 0.f, 0.f};
}

__device__ __forceinline__ bf16x4 cvt4(const f32x4 f) {
    bf16x4 b;
#pragma unroll
    for (int q = 0; q < 4; ++q) b[q] = (bf16)f[q];
    return b;
}

// ---------------------------------------------------------------------------
// Router: logits -> fp32 softmax (output #2) -> top-4 -> L1 normalize ->
// dense combine[T][E] in workspace. One wave per token.
// ---------------------------------------------------------------------------
__global__ void router_kernel(const float* __restrict__ x,
                              const float* __restrict__ rw,
                              float* __restrict__ weights_out,   // [T][E]
                              float* __restrict__ combine)       // [T][E]
{
    const int wave = threadIdx.x >> 5;
    const int lane = threadIdx.x & 31;
    const int t = blockIdx.x * 8 + wave;
    if (t >= NTOK) return;

    const float* xrow = x + (size_t)t * HID;

    float logit[NEXP];
#pragma unroll
    for (int e = 0; e < NEXP; ++e) {
        const float* wr = rw + (size_t)e * HID;
        float p = 0.f;
        for (int h = lane; h < HID; h += 32)
            p += xrow[h] * wr[h];
#pragma unroll
        for (int off = 16; off > 0; off >>= 1)
            p += __shfl_down(p, off, 32);
        logit[e] = p;   // valid in lane 0
    }

    if (lane == 0) {
        float mx = logit[0];
#pragma unroll
        for (int e = 1; e < NEXP; ++e) mx = fmaxf(mx, logit[e]);
        float w[NEXP], s = 0.f;
#pragma unroll
        for (int e = 0; e < NEXP; ++e) { w[e] = __expf(logit[e] - mx); s += w[e]; }
        const float inv = 1.f / s;
#pragma unroll
        for (int e = 0; e < NEXP; ++e) w[e] *= inv;

        float rem[NEXP], c[NEXP];
#pragma unroll
        for (int e = 0; e < NEXP; ++e) { rem[e] = w[e]; c[e] = 0.f; }
        int   te[TOPK];
        float tw[TOPK];
        float sum4 = 0.f;
#pragma unroll
        for (int k = 0; k < TOPK; ++k) {
            int best = 0; float bv = rem[0];
#pragma unroll
            for (int e = 1; e < NEXP; ++e)
                if (rem[e] > bv) { bv = rem[e]; best = e; }
            te[k] = best; tw[k] = bv; sum4 += bv;
            rem[best] = -1.f;
        }
        const float inv4 = 1.f / sum4;
#pragma unroll
        for (int k = 0; k < TOPK; ++k) c[te[k]] += tw[k] * inv4;

#pragma unroll
        for (int e = 0; e < NEXP; ++e) {
            weights_out[(size_t)t * NEXP + e] = w[e];
            combine[(size_t)t * NEXP + e]    = c[e];
        }
    }
}

// ---------------------------------------------------------------------------
// GEMM1: act[e][t][f] = silu(x @ w1[e]) * (x @ v1[e]), stored bf16.
// ---------------------------------------------------------------------------
__global__ void __launch_bounds__(256)
moe_gemm1_kernel(const float* __restrict__ x,
                 const float* __restrict__ w1,
                 const float* __restrict__ v1,
                 bf16* __restrict__ act)
{
    const int e  = blockIdx.z;
    const int t0 = blockIdx.y * TM;
    const int f0 = blockIdx.x * TN;

    __shared__ __align__(16) bf16 sx [TM][KC];   // [token][k]  (A, row-major)
    __shared__ __align__(16) bf16 swT[TN][KC];   // [f][k]      (B, transposed)
    __shared__ __align__(16) bf16 svT[TN][KC];

    const int tid  = threadIdx.x;
    const int wave = tid >> 5;
    const int lane = tid & 31;
    const int half = lane >> 4;
    const int lm   = lane & 15;
    const int nt   = wave * 16;              // wave-private N strip

    const float* w1e = w1 + (size_t)e * HID * FFN;
    const float* v1e = v1 + (size_t)e * HID * FFN;

    // static staging assignments
    const int xr = tid >> 3;                 // x tile: rows xr, xr+32
    const int xc = (tid & 7) * 4;
    const int wk = tid >> 5;                 // w tiles: k rows wk + 8*rep
    const int wc = (tid & 31) * 4;           // f cols wc..wc+3

    v8f acc_h[4], acc_v[4];
#pragma unroll
    for (int u = 0; u < 4; ++u) { acc_h[u] = v8f_zero(); acc_v[u] = v8f_zero(); }

    for (int k0 = 0; k0 < HID; k0 += KC) {
        __syncthreads();
        // ---- stage x tile (64x32 f32 -> bf16), 2 chunks/thread ----
        {
            const f32x4 fa = *reinterpret_cast<const f32x4*>(
                &x[(size_t)(t0 + xr) * HID + k0 + xc]);
            const f32x4 fb = *reinterpret_cast<const f32x4*>(
                &x[(size_t)(t0 + xr + 32) * HID + k0 + xc]);
            *reinterpret_cast<bf16x4*>(&sx[xr][xc])      = cvt4(fa);
            *reinterpret_cast<bf16x4*>(&sx[xr + 32][xc]) = cvt4(fb);
        }
        // ---- stage w1/v1 tiles (32x128 f32) transposed -> [f][k] ----
#pragma unroll
        for (int rep = 0; rep < 4; ++rep) {
            const int k = wk + rep * 8;
            const size_t gi = (size_t)(k0 + k) * FFN + f0 + wc;
            const f32x4 fw = *reinterpret_cast<const f32x4*>(&w1e[gi]);
            const f32x4 fv = *reinterpret_cast<const f32x4*>(&v1e[gi]);
#pragma unroll
            for (int q = 0; q < 4; ++q) {
                swT[wc + q][k] = (bf16)fw[q];
                svT[wc + q][k] = (bf16)fv[q];
            }
        }
        // prefetch next panel's weight rows
        if (k0 + KC < HID) {
            const size_t pn = (size_t)(k0 + KC + wk) * FFN + f0 + wc;
            __builtin_prefetch(&w1e[pn], 0, 1);
            __builtin_prefetch(&v1e[pn], 0, 1);
        }
        __syncthreads();

        // ---- compute: B loaded once per wave, reused over 4 M sub-tiles ----
        const v16bf Bw = frag_b(&swT[nt + lm][0], half);
        const v16bf Bv = frag_b(&svT[nt + lm][0], half);
#pragma unroll
        for (int u = 0; u < 4; ++u) {
            const v16bf A = frag_a(&sx[u * 16 + lm][0], half);
            acc_h[u] = __builtin_amdgcn_wmma_f32_16x16x32_bf16(
                false, A, false, Bw, (short)0, acc_h[u], false, false);
            acc_v[u] = __builtin_amdgcn_wmma_f32_16x16x32_bf16(
                false, A, false, Bv, (short)0, acc_v[u], false, false);
        }
    }

    // ---- epilogue: silu(h) * v -> bf16 activation ----
#pragma unroll
    for (int u = 0; u < 4; ++u) {
#pragma unroll
        for (int r = 0; r < 8; ++r) {
            const int m = u * 16 + r + half * 8;     // C/D layout: M = r + 8*half
            const float hv  = acc_h[u][r];
            const float sig = 1.f / (1.f + __expf(-hv));
            const float a   = hv * sig * acc_v[u][r];
            act[((size_t)e * NTOK + (t0 + m)) * FFN + f0 + nt + lm] = (bf16)a;
        }
    }
}

// ---------------------------------------------------------------------------
// GEMM2: out[t][h] = sum_e combine[t][e] * (act[e] @ w2[e])[t][h].
// Experts looped inside the block -> no atomics on out.
// ---------------------------------------------------------------------------
__global__ void __launch_bounds__(256)
moe_gemm2_kernel(const bf16* __restrict__ act,
                 const float* __restrict__ w2,
                 const float* __restrict__ combine,
                 float* __restrict__ out)
{
    const int h0 = blockIdx.x * TN;
    const int t0 = blockIdx.y * TM;

    __shared__ __align__(16) bf16  sa [TM][KC];    // [token][k] (A)
    __shared__ __align__(16) bf16  sbT[TN][KC];    // [h][k]     (B transposed)
    __shared__ __align__(16) float sc [TM][NEXP];  // combine tile

    const int tid  = threadIdx.x;
    const int wave = tid >> 5;
    const int lane = tid & 31;
    const int half = lane >> 4;
    const int lm   = lane & 15;
    const int nt   = wave * 16;

    // static staging assignments
    const int ar = tid >> 2;                 // sa: row ar, 8-elem chunk
    const int ac = (tid & 3) * 8;
    const int wk = tid >> 5;                 // sbT: k rows wk + 8*rep
    const int wc = (tid & 31) * 4;           // h cols wc..wc+3

    for (int i = tid; i < TM * NEXP; i += 256)
        sc[i >> 4][i & 15] = combine[(size_t)(t0 + (i >> 4)) * NEXP + (i & 15)];

    v8f out_acc[4];
#pragma unroll
    for (int u = 0; u < 4; ++u) out_acc[u] = v8f_zero();

    for (int e = 0; e < NEXP; ++e) {
        const bf16*  acte = act + (size_t)e * NTOK * FFN;
        const float* w2e  = w2  + (size_t)e * FFN * HID;

        v8f acc[4];
#pragma unroll
        for (int u = 0; u < 4; ++u) acc[u] = v8f_zero();

        for (int k0 = 0; k0 < FFN; k0 += KC) {
            __syncthreads();
            // sa: act already bf16 -> pure 16B copy, 1 chunk/thread
            *reinterpret_cast<bf16x8*>(&sa[ar][ac]) =
                *reinterpret_cast<const bf16x8*>(
                    &acte[(size_t)(t0 + ar) * FFN + k0 + ac]);
            // sbT: w2 (32x128 f32) transposed -> [h][k]
#pragma unroll
            for (int rep = 0; rep < 4; ++rep) {
                const int k = wk + rep * 8;
                const f32x4 fw = *reinterpret_cast<const f32x4*>(
                    &w2e[(size_t)(k0 + k) * HID + h0 + wc]);
#pragma unroll
                for (int q = 0; q < 4; ++q)
                    sbT[wc + q][k] = (bf16)fw[q];
            }
            if (k0 + KC < FFN) {
                __builtin_prefetch(&w2e[(size_t)(k0 + KC + wk) * HID + h0 + wc], 0, 1);
                __builtin_prefetch(&acte[(size_t)(t0 + ar) * FFN + k0 + KC + ac], 0, 1);
            }
            __syncthreads();

            const v16bf B = frag_b(&sbT[nt + lm][0], half);
#pragma unroll
            for (int u = 0; u < 4; ++u) {
                const v16bf A = frag_a(&sa[u * 16 + lm][0], half);
                acc[u] = __builtin_amdgcn_wmma_f32_16x16x32_bf16(
                    false, A, false, B, (short)0, acc[u], false, false);
            }
        }

        // combine-weighted accumulation (per-token scale)
#pragma unroll
        for (int u = 0; u < 4; ++u) {
#pragma unroll
            for (int r = 0; r < 8; ++r) {
                const int m = u * 16 + r + half * 8;
                out_acc[u][r] += sc[m][e] * acc[u][r];
            }
        }
    }

#pragma unroll
    for (int u = 0; u < 4; ++u) {
#pragma unroll
        for (int r = 0; r < 8; ++r) {
            const int m = u * 16 + r + half * 8;
            out[(size_t)(t0 + m) * HID + h0 + nt + lm] = out_acc[u][r];
        }
    }
}

// ---------------------------------------------------------------------------
// Launch
// ---------------------------------------------------------------------------
extern "C" void kernel_launch(void* const* d_in, const int* in_sizes, int n_in,
                              void* d_out, int out_size, void* d_ws, size_t ws_size,
                              hipStream_t stream)
{
    (void)in_sizes; (void)n_in; (void)out_size; (void)ws_size;

    const float* x        = (const float*)d_in[0];   // [1,2048,2048]
    const float* router_w = (const float*)d_in[1];   // [16,2048]
    const float* w1       = (const float*)d_in[2];   // [16,2048,4096]
    const float* v1       = (const float*)d_in[3];   // [16,2048,4096]
    const float* w2       = (const float*)d_in[4];   // [16,4096,2048]

    float* out         = (float*)d_out;                          // [T*H]
    float* weights_out = out + (size_t)NTOK * HID;               // [T*E]

    // workspace: combine [T*E] f32, then act [E*T*F] bf16
    float* combine = (float*)d_ws;
    bf16*  act     = (bf16*)((char*)d_ws + (size_t)NTOK * NEXP * sizeof(float));

    router_kernel<<<dim3(NTOK / 8), dim3(256), 0, stream>>>(
        x, router_w, weights_out, combine);

    moe_gemm1_kernel<<<dim3(FFN / TN, NTOK / TM, NEXP), dim3(256), 0, stream>>>(
        x, w1, v1, act);

    moe_gemm2_kernel<<<dim3(HID / TN, NTOK / TM), dim3(256), 0, stream>>>(
        act, w2, combine, out);
}
